// VariableSelectionNetwork_53231824666759
// MI455X (gfx1250) — compile-verified, matrix operand order checked by reading, not run on hardware
//
#include <hip/hip_runtime.h>

typedef __attribute__((ext_vector_type(16))) __bf16       v16bf;
typedef __attribute__((ext_vector_type(8)))  float        v8f;
typedef __attribute__((ext_vector_type(4)))  unsigned int v4u;

#define NB   4096      // batch
#define NF   64        // features
#define HD   256       // hidden
#define BM   64        // batch rows per block (kernel 2)
#define NGRP 8         // features per block (kernel 2)

// ---------------- helpers ----------------

// XOR swizzle on 8-dword chunks so lanes l and l+8 (which collide at stride
// 8 dwords) land in different bank groups; 16B alignment preserved.
__device__ __forceinline__ unsigned swz8(unsigned ci) {
    unsigned off = ci * 8u;                       // dword offset
    return off ^ (((off >> 6) & 7u) << 3);
}

// Split f32 into bf16 hi + bf16 lo (truncation; lo captures the residual).
__device__ __forceinline__ void split_bf(float v, unsigned short& hi, unsigned short& lo) {
    unsigned uv = __float_as_uint(v);
    hi = (unsigned short)(uv >> 16);
    float lf = v - __uint_as_float(uv & 0xFFFF0000u);
    lo = (unsigned short)(__float_as_uint(lf) >> 16);
}

// Build a v16bf WMMA operand from two 16B LDS chunks (2x ds_load_b128).
__device__ __forceinline__ v16bf load_op_split(const unsigned short* p0, const unsigned short* p1) {
    union { v4u q[2]; v16bf v; } u;
    u.q[0] = *(const v4u*)p0;
    u.q[1] = *(const v4u*)p1;
    return u.v;
}

// ---------------- kernel 1: sparse_w = softmax(LN(GRN(x))) ----------------
// one block (64 threads = 2 waves) per batch row; also zeroes combined region.

__global__ __launch_bounds__(64) void vsn_weights(
    const float* __restrict__ x,
    const float* __restrict__ f1w, const float* __restrict__ f1b,
    const float* __restrict__ f2w, const float* __restrict__ f2b,
    const float* __restrict__ skw, const float* __restrict__ skb,
    const float* __restrict__ gtw, const float* __restrict__ gtb,
    const float* __restrict__ lng, const float* __restrict__ lnb,
    float* __restrict__ out)
{
    __shared__ float xr[64];
    __shared__ float ts[256];
    __shared__ float red[64];
    const int t = threadIdx.x;
    const int r = blockIdx.x;

    xr[t] = x[(size_t)r * NF + t];
    // zero the combined output region (atomic-accumulated by kernel 2)
    for (int q2 = 0; q2 < 4; ++q2) out[(size_t)r * HD + q2 * 64 + t] = 0.0f;
    __syncthreads();

    // h = elu(x @ f_fc1_w + b)   (64 -> 256)
    for (int q2 = 0; q2 < 4; ++q2) {
        int h = q2 * 64 + t;
        float a = f1b[h];
        for (int i = 0; i < 64; ++i) a = fmaf(xr[i], f1w[i * HD + h], a);
        ts[h] = a > 0.0f ? a : (__expf(a) - 1.0f);
    }
    __syncthreads();

    // fc2 (256 -> 64), skip, gate  (all coalesced down columns)
    float h2 = f2b[t];
    for (int i = 0; i < 256; ++i) h2 = fmaf(ts[i], f2w[i * 64 + t], h2);
    float sk = skb[t], z = gtb[t];
    for (int i = 0; i < 64; ++i) {
        sk = fmaf(xr[i], skw[i * 64 + t], sk);
        z  = fmaf(xr[i], gtw[i * 64 + t], z);
    }
    float g = 1.0f / (1.0f + __expf(-z));
    float u = g * h2 + (1.0f - g) * sk;

    // LayerNorm over 64
    red[t] = u;
    __syncthreads();
    float mu = 0.0f, ms = 0.0f;
    for (int i = 0; i < 64; ++i) { mu += red[i]; ms += red[i] * red[i]; }
    mu *= (1.0f / 64.0f); ms *= (1.0f / 64.0f);
    float wv = fmaf((u - mu) * rsqrtf(ms - mu * mu + 1e-5f), lng[t], lnb[t]);
    __syncthreads();

    // softmax over 64
    red[t] = wv;
    __syncthreads();
    float mx = -3.0e38f;
    for (int i = 0; i < 64; ++i) mx = fmaxf(mx, red[i]);
    float e = __expf(wv - mx);
    __syncthreads();
    red[t] = e;
    __syncthreads();
    float se = 0.0f;
    for (int i = 0; i < 64; ++i) se += red[i];

    out[(size_t)NB * HD + (size_t)r * NF + t] = e / se;
}

// ---------------- kernel 2: per-feature GRN einsum + LN + weighted sum ----
// grid = (NB/BM, NF/NGRP); 256 threads = 8 wave32.
// wave w: m-tile = w>>1 (16 rows), n-half = w&1 (128 cols) -> 8 C tiles.
// W slabs are register-prefetched and the LDS pack buffer is double-buffered
// so the k-loop runs with one barrier per step and global latency hidden.

__global__ __launch_bounds__(256) void vsn_embed(
    const float* __restrict__ x,
    const float* __restrict__ g1w, const float* __restrict__ g1b,
    const float* __restrict__ g2w, const float* __restrict__ g2b,
    const float* __restrict__ skw, const float* __restrict__ skb,
    const float* __restrict__ gtw, const float* __restrict__ gtb,
    const float* __restrict__ lng, const float* __restrict__ lnb,
    const float* __restrict__ sw,  float* __restrict__ out)
{
    extern __shared__ char smem[];
    unsigned short* aHi = (unsigned short*)smem;        // 64 x 264 bf16 (A hi)
    unsigned short* aLo = aHi + 64 * 264;               // 64 x 264 bf16 (A lo)
    unsigned short* wpk = aLo + 64 * 264;               // 2 buffers x (hi[8192]+lo[8192])
    float* comb = (float*)(wpk + 2 * 16384);            // 64 x 256 f32 accum
    float* epi  = comb + BM * HD;                       // 7 x 256 epilogue weights
    float* xs   = epi + 7 * HD;                         // 8 x 64 x values
    float* sws  = xs + NGRP * 64;                       // 8 x 64 softmax weights
    float* rred = sws + NGRP * 64;                      // 64 x 4 row stats partials

    const int tid  = threadIdx.x;
    const int lane = tid & 31;
    const int wvid = tid >> 5;
    const int mt   = wvid >> 1;       // 0..3
    const int nh   = wvid & 1;        // 0..1
    const int l15  = lane & 15;
    const int lh   = lane >> 4;
    const int row0 = blockIdx.x * BM;
    const int ng0  = blockIdx.y * NGRP;

    for (int i = tid; i < NGRP * 64; i += 256) {
        int j = i >> 6, r = i & 63;
        xs[i]  = x [(size_t)(row0 + r) * NF + ng0 + j];
        sws[i] = sw[(size_t)(row0 + r) * NF + ng0 + j];
    }
    for (int i = tid; i < BM * HD; i += 256) comb[i] = 0.0f;
    __syncthreads();

    for (int j = 0; j < NGRP; ++j) {
        const int n = ng0 + j;
        const float* Wn = g2w + (size_t)n * HD * HD;

        // ---- prefetch W slab 0 into registers (overlaps A staging below)
        float vbuf[32];
        #pragma unroll
        for (int i = 0; i < 32; ++i) vbuf[i] = Wn[i * HD + tid];

        // ---- stage A = elu(x*w1 + b1) as bf16 hi/lo; stage epilogue weights
        {
            float w1 = g1w[n * HD + tid];
            float b1 = g1b[n * HD + tid];
            for (int r = 0; r < BM; ++r) {
                float v = fmaf(xs[j * 64 + r], w1, b1);
                v = v > 0.0f ? v : (__expf(v) - 1.0f);
                unsigned short h, l;
                split_bf(v, h, l);
                aHi[r * 264 + tid] = h;
                aLo[r * 264 + tid] = l;
            }
            epi[0 * HD + tid] = g2b[n * HD + tid];
            epi[1 * HD + tid] = skw[n * HD + tid];
            epi[2 * HD + tid] = skb[n * HD + tid];
            epi[3 * HD + tid] = gtw[n * HD + tid];
            epi[4 * HD + tid] = gtb[n * HD + tid];
            epi[5 * HD + tid] = lng[n * HD + tid];
            epi[6 * HD + tid] = lnb[n * HD + tid];
        }

        v8f acc[8];
        #pragma unroll
        for (int t = 0; t < 8; ++t) {
            #pragma unroll
            for (int e = 0; e < 8; ++e) acc[t][e] = 0.0f;
        }

        for (int k = 0; k < 8; ++k) {
            // ---- pack prefetched slab into B-operand layout, buffer k&1
            unsigned short* wb = wpk + (k & 1) * 16384;
            #pragma unroll
            for (int i = 0; i < 32; ++i) {
                unsigned short h, l;
                split_bf(vbuf[i], h, l);
                unsigned ci = (unsigned)((tid >> 4) * 32 + (tid & 15) + 16 * ((i >> 3) & 1));
                unsigned us = swz8(ci) * 2u + (unsigned)((i & 7) + 8 * (i >> 4));
                wb[us]        = h;
                wb[8192 + us] = l;
            }
            // ---- prefetch next slab (completes under the WMMA phase)
            if (k < 7) {
                const float* Ws = Wn + (k + 1) * 32 * HD;
                #pragma unroll
                for (int i = 0; i < 32; ++i) vbuf[i] = Ws[i * HD + tid];
            }
            __syncthreads();
            // ---- WMMA: acc += Ahi*Bhi + Ahi*Blo + Alo*Bhi
            const int arow = mt * 16 + l15;
            const int klo  = k * 32 + lh * 8;
            const unsigned short* pa = aHi + arow * 264 + klo;
            const unsigned short* pl = aLo + arow * 264 + klo;
            v16bf ahi = load_op_split(pa, pa + 16);
            v16bf alo = load_op_split(pl, pl + 16);
            #pragma unroll
            for (int t = 0; t < 8; ++t) {
                unsigned ci = (unsigned)((nh * 8 + t) * 32 + lane);
                const unsigned short* pb = wb + swz8(ci) * 2u;
                v16bf bhi = load_op_split(pb, pb + 8);
                v16bf blo = load_op_split(pb + 8192, pb + 8192 + 8);
                acc[t] = __builtin_amdgcn_wmma_f32_16x16x32_bf16(false, ahi, false, bhi, (short)0, acc[t], false, false);
                acc[t] = __builtin_amdgcn_wmma_f32_16x16x32_bf16(false, ahi, false, blo, (short)0, acc[t], false, false);
                acc[t] = __builtin_amdgcn_wmma_f32_16x16x32_bf16(false, alo, false, bhi, (short)0, acc[t], false, false);
            }
        }

        // ---- epilogue pass 1: gate/skip fuse, row-sum partials
        #pragma unroll
        for (int jv = 0; jv < 8; ++jv) {
            const int rl = mt * 16 + jv + 8 * lh;     // local row 0..63
            const float xv = xs[j * 64 + rl];
            float s = 0.0f, q = 0.0f;
            #pragma unroll
            for (int t = 0; t < 8; ++t) {
                const int c = (nh * 8 + t) * 16 + l15;
                float h2 = acc[t][jv] + epi[0 * HD + c];
                float sk = fmaf(xv, epi[1 * HD + c], epi[2 * HD + c]);
                float z  = fmaf(xv, epi[3 * HD + c], epi[4 * HD + c]);
                float g  = 1.0f / (1.0f + __expf(-z));
                float u  = g * h2 + (1.0f - g) * sk;
                acc[t][jv] = u;
                s += u; q += u * u;
            }
            s += __shfl_xor(s, 1, 32); s += __shfl_xor(s, 2, 32);
            s += __shfl_xor(s, 4, 32); s += __shfl_xor(s, 8, 32);
            q += __shfl_xor(q, 1, 32); q += __shfl_xor(q, 2, 32);
            q += __shfl_xor(q, 4, 32); q += __shfl_xor(q, 8, 32);
            if (l15 == 0) { rred[rl * 4 + nh] = s; rred[rl * 4 + 2 + nh] = q; }
        }
        __syncthreads();

        // ---- epilogue pass 2: LayerNorm + softmax-weighted accumulate
        #pragma unroll
        for (int jv = 0; jv < 8; ++jv) {
            const int rl = mt * 16 + jv + 8 * lh;
            float mu = (rred[rl * 4 + 0] + rred[rl * 4 + 1]) * (1.0f / 256.0f);
            float ms = (rred[rl * 4 + 2] + rred[rl * 4 + 3]) * (1.0f / 256.0f);
            float rs = rsqrtf(ms - mu * mu + 1e-5f);
            float swv = sws[j * 64 + rl];
            #pragma unroll
            for (int t = 0; t < 8; ++t) {
                const int c = (nh * 8 + t) * 16 + l15;
                float emb = fmaf((acc[t][jv] - mu) * rs, epi[5 * HD + c], epi[6 * HD + c]);
                comb[rl * HD + c] += swv * emb;       // owner-exclusive, no race
            }
        }
        __syncthreads();
    }

    // ---- flush block-local accumulation (n-groups combine via f32 atomics)
    for (int i = tid; i < BM * HD; i += 256) {
        int r = i >> 8, c = i & 255;
        atomicAdd(&out[(size_t)(row0 + r) * HD + c], comb[i]);
    }
}

// ---------------- launcher ----------------

extern "C" void kernel_launch(void* const* d_in, const int* in_sizes, int n_in,
                              void* d_out, int out_size, void* d_ws, size_t ws_size,
                              hipStream_t stream) {
    (void)in_sizes; (void)n_in; (void)d_ws; (void)ws_size; (void)out_size;

    const float* x   = (const float*)d_in[0];
    const float* f1w = (const float*)d_in[1];
    const float* f1b = (const float*)d_in[2];
    const float* f2w = (const float*)d_in[3];
    const float* f2b = (const float*)d_in[4];
    const float* fsw = (const float*)d_in[5];
    const float* fsb = (const float*)d_in[6];
    const float* fgw = (const float*)d_in[7];
    const float* fgb = (const float*)d_in[8];
    const float* flg = (const float*)d_in[9];
    const float* flb = (const float*)d_in[10];
    const float* g1w = (const float*)d_in[11];
    const float* g1b = (const float*)d_in[12];
    const float* g2w = (const float*)d_in[13];
    const float* g2b = (const float*)d_in[14];
    const float* gsw = (const float*)d_in[15];
    const float* gsb = (const float*)d_in[16];
    const float* ggw = (const float*)d_in[17];
    const float* ggb = (const float*)d_in[18];
    const float* glg = (const float*)d_in[19];
    const float* glb = (const float*)d_in[20];

    float* out = (float*)d_out;

    // phase 1: sparse_w into out[NB*HD ..], zero combined region
    vsn_weights<<<dim3(NB), dim3(64), 0, stream>>>(
        x, f1w, f1b, f2w, f2b, fsw, fsb, fgw, fgb, flg, flb, out);

    const float* sparse = out + (size_t)NB * HD;

    // dynamic LDS: A hi/lo (67584) + W pack double-buffer (65536)
    //            + comb (65536) + epi (7168) + xs/sws (4096) + rred (1024)
    //            = 210944 B  (fits 320KB WGP LDS)
    size_t lds = 210944;
    vsn_embed<<<dim3(NB / BM, NF / NGRP), dim3(256), lds, stream>>>(
        x, g1w, g1b, g2w, g2b, gsw, gsb, ggw, ggb, glg, glb, sparse, out);
}
